// DCIRNeuralODE_13039520710746
// MI455X (gfx1250) — matrix operand order repaired and checked by reading, not compiled, not run on hardware
//
#include <hip/hip_runtime.h>

typedef __attribute__((ext_vector_type(16))) _Float16 v16h;
typedef __attribute__((ext_vector_type(8)))  float    v8f;

#define BATCH  32768
#define HSTEPS 256
#define HID    128
#define WAVES  4
#define TPB    (WAVES * 32)
#define EPSF   1e-6f
#define QINV   (1.0f / (3.0f * 3600.0f))

__device__ __forceinline__ float fast_tanh(float x) {
#if __has_builtin(__builtin_amdgcn_tanhf)
    return __builtin_amdgcn_tanhf(x);        // native v_tanh_f32 (CDNA5 TRANS op)
#else
    // tanh(x) = 1 - 2/(exp(2x)+1); saturates correctly at +-1
    float e = __expf(2.0f * x);
    float r = __builtin_amdgcn_rcpf(e + 1.0f);
    return 1.0f - 2.0f * r;
#endif
}
__device__ __forceinline__ float fast_softplus(float x) {
    return __logf(1.0f + __expf(x));
}

// Intra-wave LDS exchange fence: LDS ops of one wave execute in program
// order (DScnt), so cross-lane visibility within a wave only requires
// blocking compiler reordering — no block-wide s_barrier needed.
__device__ __forceinline__ void wave_lds_fence() {
    __asm__ volatile("" ::: "memory");
    __builtin_amdgcn_wave_barrier();
    __asm__ volatile("" ::: "memory");
}

__global__ __launch_bounds__(TPB) void dcir_neural_ode_kernel(
    const float* __restrict__ Ig,  const float* __restrict__ Tg,
    const float* __restrict__ soc0, const float* __restrict__ W1,
    const float* __restrict__ b1,  const float* __restrict__ W2,
    const float* __restrict__ b2,  const float* __restrict__ Wr1,
    const float* __restrict__ br1, const float* __restrict__ Wr2,
    const float* __restrict__ br2, float* __restrict__ out)
{
    __shared__ __align__(16) float sW0[HID];   // W1 row 0
    __shared__ __align__(16) float sW1r[HID];  // W1 row 1
    __shared__ __align__(16) float sB1[HID];   // b1
    __shared__ float sP[WAVES][16 * 5];        // params exchange (D-layout -> lane)
    __shared__ float sY[WAVES][16];            // residual scalar exchange

    const int tid  = threadIdx.x;
    if (tid < HID) { sW0[tid] = W1[tid]; sW1r[tid] = W1[HID + tid]; sB1[tid] = b1[tid]; }

    const int lane = tid & 31;
    const int w    = tid >> 5;
    const int row  = lane & 15;   // battery within tile; also N column in B/C/D layouts
    const int s    = lane >> 4;   // half-wave index
    const int base = (blockIdx.x * WAVES + w) * 16;
    const int bat  = base + row;

    // ---- Step-invariant WMMA B fragments (16-bit B 32x16 layout:
    //      lane holds element (K = 8*s + j (j<8) or 8*s + 8 + j (j>=8), N = row) in half j) ----
    // Layer2: W2 padded [128,16], split into 4 K-chunks of 32.
    v16h bw2[4];
    #pragma unroll
    for (int c = 0; c < 4; ++c) {
        v16h bb = {};
        #pragma unroll
        for (int j = 0; j < 16; ++j) {
            int kk = 32 * c + 8 * s + j + ((j >= 8) ? 8 : 0);
            bb[j] = (row < 5) ? (_Float16)W2[kk * 5 + row] : (_Float16)0.0f;
        }
        bw2[c] = bb;
    }
    // Residual layer1: Wr1 padded [32,64] -> 4 N-tiles of [32,16]; only K<5 nonzero.
    v16h brf[4];
    #pragma unroll
    for (int t = 0; t < 4; ++t) {
        v16h bb = {};
        if (s == 0) {
            #pragma unroll
            for (int j = 0; j < 5; ++j) bb[j] = (_Float16)Wr1[j * 64 + 16 * t + row];
        }
        brf[t] = bb;
    }
    float w2v[4], rb1[4];
    #pragma unroll
    for (int t = 0; t < 4; ++t) { w2v[t] = Wr2[16 * t + row]; rb1[t] = br1[16 * t + row]; }

    const float accInit = (row < 5) ? b2[row] : 0.0f;
    const float sc   = (row == 2) ? 2000.0f : ((row == 4) ? 10000.0f : 0.005f);
    const float br2v = br2[0];

    float vc1 = 0.0f, vc2 = 0.0f, soc = soc0[bat];

    const float* Irow = Ig + (size_t)bat * HSTEPS;
    const float* Trow = Tg + (size_t)bat * HSTEPS;
    float* Vout = out;                                   // [B,H]
    float* Pout = out + (size_t)BATCH * HSTEPS;          // [B,H,5]
    float* Sout = out + (size_t)BATCH * HSTEPS * 6;      // [B,H,3]

    __syncthreads();   // weights staged to LDS (cross-wave dependency)

    #pragma unroll 1
    for (int k = 0; k < HSTEPS; ++k) {
        if ((k & 15) == 0 && k + 16 < HSTEPS) {
            __builtin_prefetch(Irow + k + 16, 0, 3);     // global_prefetch_b8
            __builtin_prefetch(Trow + k + 16, 0, 3);
        }
        const float iv = Irow[k];
        const float tv = Trow[k];

        // ================= param head =================
        // h = tanh([soc,t] @ W1 + b1) computed lane-local directly in the
        // 16-bit A 16x32 fragment layout: half j of chunk c holds
        // k = 32c + 8s + j (j<8) or 32c + 8s + 8 + j (j>=8); row M = lane%16.
        v8f acc;
        #pragma unroll
        for (int r = 0; r < 8; ++r) acc[r] = accInit;

        #pragma unroll
        for (int c = 0; c < 4; ++c) {
            const int k0 = 32 * c + 8 * s;
            float4 A0 = *(const float4*)&sW0[k0];
            float4 A1 = *(const float4*)&sW0[k0 + 4];
            float4 A2 = *(const float4*)&sW0[k0 + 16];
            float4 A3 = *(const float4*)&sW0[k0 + 20];
            float4 B0 = *(const float4*)&sW1r[k0];
            float4 B1 = *(const float4*)&sW1r[k0 + 4];
            float4 B2 = *(const float4*)&sW1r[k0 + 16];
            float4 B3 = *(const float4*)&sW1r[k0 + 20];
            float4 C0 = *(const float4*)&sB1[k0];
            float4 C1v = *(const float4*)&sB1[k0 + 4];
            float4 C2v = *(const float4*)&sB1[k0 + 16];
            float4 C3 = *(const float4*)&sB1[k0 + 20];

            v16h a = {};
            #define H1(idx, wa, wb, bb) \
                a[idx] = (_Float16)fast_tanh(fmaf(soc, wa, fmaf(tv, wb, bb)));
            H1(0,  A0.x, B0.x, C0.x)  H1(1,  A0.y, B0.y, C0.y)
            H1(2,  A0.z, B0.z, C0.z)  H1(3,  A0.w, B0.w, C0.w)
            H1(4,  A1.x, B1.x, C1v.x) H1(5,  A1.y, B1.y, C1v.y)
            H1(6,  A1.z, B1.z, C1v.z) H1(7,  A1.w, B1.w, C1v.w)
            H1(8,  A2.x, B2.x, C2v.x) H1(9,  A2.y, B2.y, C2v.y)
            H1(10, A2.z, B2.z, C2v.z) H1(11, A2.w, B2.w, C2v.w)
            H1(12, A3.x, B3.x, C3.x)  H1(13, A3.y, B3.y, C3.y)
            H1(14, A3.z, B3.z, C3.z)  H1(15, A3.w, B3.w, C3.w)
            #undef H1

            acc = __builtin_amdgcn_wmma_f32_16x16x32_f16(
                false, a, false, bw2[c], (short)0, acc, false, false);
        }

        // softplus + scale, stash params to LDS (D layout: lane holds
        // element (M = r + 8s, N = row) in acc[r])
        #pragma unroll
        for (int r = 0; r < 8; ++r) {
            float pv = fmaf(fast_softplus(acc[r]), sc, EPSF);
            if (row < 5) sP[w][(r + 8 * s) * 5 + row] = pv;
        }

        // ================= residual MLP =================
        // A = [vc1,vc2,soc,i,t] padded 16x32: only k<5 nonzero (s==0, j<5)
        v16h ar = {};
        if (s == 0) {
            ar[0] = (_Float16)vc1; ar[1] = (_Float16)vc2; ar[2] = (_Float16)soc;
            ar[3] = (_Float16)iv;  ar[4] = (_Float16)tv;
        }
        float part[8];
        #pragma unroll
        for (int r = 0; r < 8; ++r) part[r] = 0.0f;
        #pragma unroll
        for (int t = 0; t < 4; ++t) {
            v8f rc;
            #pragma unroll
            for (int r = 0; r < 8; ++r) rc[r] = rb1[t];
            rc = __builtin_amdgcn_wmma_f32_16x16x32_f16(
                false, ar, false, brf[t], (short)0, rc, false, false);
            #pragma unroll
            for (int r = 0; r < 8; ++r)
                part[r] = fmaf(fast_tanh(rc[r]), w2v[t], part[r]);
        }
        // reduce over the 16 N-lanes of each half-wave (butterfly)
        #pragma unroll
        for (int m = 1; m < 16; m <<= 1) {
            #pragma unroll
            for (int r = 0; r < 8; ++r) part[r] += __shfl_xor(part[r], m, 32);
        }
        if (row == 0) {
            #pragma unroll
            for (int r = 0; r < 8; ++r) sY[w][r + 8 * s] = part[r];
        }
        wave_lds_fence();   // per-wave scratch: compiler fence only, no s_barrier

        // ================= per-battery scalar path =================
        const float R0 = sP[w][row * 5 + 0];
        const float R1 = sP[w][row * 5 + 1];
        const float C1 = sP[w][row * 5 + 2];
        const float R2 = sP[w][row * 5 + 3];
        const float C2 = sP[w][row * 5 + 4];
        const float yres = sY[w][row];

        const float ocv  = 3.0f + 1.2f * soc - 0.4f * __expf(-12.0f * soc);
        const float Vecm = ocv - R0 * iv - vc1 - vc2;
        const float Vp   = Vecm + (yres + br2v) * 0.01f;

        if (s == 0) {
            const size_t o  = (size_t)bat * HSTEPS + k;
            Vout[o] = Vp;
            const size_t po = o * 5;
            Pout[po + 0] = R0; Pout[po + 1] = R1; Pout[po + 2] = C1;
            Pout[po + 3] = R2; Pout[po + 4] = C2;
            const size_t so = o * 3;
            Sout[so + 0] = vc1; Sout[so + 1] = vc2; Sout[so + 2] = soc;
        }

        // ================= RK4 update (DT = 1) =================
        const float a1 = __builtin_amdgcn_rcpf(R1 * C1);
        const float u1 = iv * __builtin_amdgcn_rcpf(C1);
        const float a2 = __builtin_amdgcn_rcpf(R2 * C2);
        const float u2 = iv * __builtin_amdgcn_rcpf(C2);
        {
            float k1 = u1 - a1 * vc1;
            float k2 = u1 - a1 * fmaf(0.5f, k1, vc1);
            float k3 = u1 - a1 * fmaf(0.5f, k2, vc1);
            float k4 = u1 - a1 * (vc1 + k3);
            vc1 = vc1 + (k1 + 2.0f * k2 + 2.0f * k3 + k4) * (1.0f / 6.0f);
        }
        {
            float k1 = u2 - a2 * vc2;
            float k2 = u2 - a2 * fmaf(0.5f, k1, vc2);
            float k3 = u2 - a2 * fmaf(0.5f, k2, vc2);
            float k4 = u2 - a2 * (vc2 + k3);
            vc2 = vc2 + (k1 + 2.0f * k2 + 2.0f * k3 + k4) * (1.0f / 6.0f);
        }
        soc = soc - iv * QINV;
        soc = fminf(fmaxf(soc, 0.0f), 1.0f);

        wave_lds_fence();   // protect sP/sY before next step overwrites
    }
}

extern "C" void kernel_launch(void* const* d_in, const int* in_sizes, int n_in,
                              void* d_out, int out_size, void* d_ws, size_t ws_size,
                              hipStream_t stream) {
    (void)in_sizes; (void)n_in; (void)out_size; (void)d_ws; (void)ws_size;
    // d_in order: V, I, Tz, soc0, W1, b1, W2, b2, Wr1, br1, Wr2, br2 (V unused)
    const float* Ig   = (const float*)d_in[1];
    const float* Tg   = (const float*)d_in[2];
    const float* soc0 = (const float*)d_in[3];
    const float* W1   = (const float*)d_in[4];
    const float* b1   = (const float*)d_in[5];
    const float* W2   = (const float*)d_in[6];
    const float* b2   = (const float*)d_in[7];
    const float* Wr1  = (const float*)d_in[8];
    const float* br1  = (const float*)d_in[9];
    const float* Wr2  = (const float*)d_in[10];
    const float* br2  = (const float*)d_in[11];

    dim3 grid(BATCH / (16 * WAVES));
    dcir_neural_ode_kernel<<<grid, TPB, 0, stream>>>(
        Ig, Tg, soc0, W1, b1, W2, b2, Wr1, br1, Wr2, br2, (float*)d_out);
}
